// SpatialNeighbourhoodAttentionBlock_24541443130184
// MI455X (gfx1250) — compile-verified, matrix operand order checked by reading, not run on hardware
//
#include <hip/hip_runtime.h>
#include <hip/hip_bf16.h>

// ---------------------------------------------------------------------------
// Spatial neighborhood attention block (B=2, C=384, T=4, H=W=32, K=7)
//   M = B*T*H*W = 8192 rows, Cin = 384, QKV N = 1152, proj N = 384
// GEMMs: v_wmma_f32_16x16x32_bf16, register double-buffered mainloop.
// Attention: wave-per-pixel, async global->LDS q staging, shfl softmax.
// ---------------------------------------------------------------------------

typedef __attribute__((ext_vector_type(16))) __bf16 v16bf;
typedef __attribute__((ext_vector_type(2)))  __bf16 v2bf;
typedef __attribute__((ext_vector_type(8)))  float  v8f;

#define CC     384
#define HH     32
#define WW     32
#define TT     4
#define HWSZ   1024            // H*W
#define MROWS  8192            // B*T*H*W
#define N_QKV  1152
#define KWIN   7
#define QSCALE 0.05103103630798288f   // 384^-0.5
#define SQRT_C 19.595917942265423f    // sqrt(384)

__device__ __forceinline__ unsigned short f32_to_bf16_rne(float f) {
    unsigned int u = __builtin_bit_cast(unsigned int, f);
    unsigned int r = (u + 0x7FFFu + ((u >> 16) & 1u)) >> 16;
    return (unsigned short)r;
}

__device__ __forceinline__ int iclamp(int v, int lo, int hi) {
    return v < lo ? lo : (v > hi ? hi : v);
}

// Async global -> LDS copy of 16 bytes (ASYNCcnt-tracked, CDNA5 §15.18.3).
__device__ __forceinline__ void async_copy_b128(void* lds_ptr, const void* gptr) {
    unsigned int lds_off = (unsigned int)(size_t)lds_ptr;   // addr[31:0] = LDS offset
    asm volatile("global_load_async_to_lds_b128 %0, %1, off"
                 :: "v"(lds_off), "v"(gptr) : "memory");
}
__device__ __forceinline__ void wait_asynccnt0() {
    asm volatile("s_wait_asynccnt 0x0" ::: "memory");
}

// Load a 16x32 bf16 fragment (A or B operand) from a row-major matrix with
// K contiguous (leading dim = ld).  CDNA5 wave32 layout:
//   lane L: row = L & 15, K-half = L >> 4
//   VGPR r (element pair 2r,2r+1): K = (r>=4 ? 16 : 0) + 8*Khalf + (r&3)*2
// (Compiler folds khalf into the base address -> two global_load_b128.)
__device__ __forceinline__ v16bf load_frag16x32(const unsigned short* __restrict__ base,
                                                int ld, int row0, int k0, int lane) {
    const int m = lane & 15;
    const int khalf = (lane >> 4) & 1;
    const unsigned short* p = base + (size_t)(row0 + m) * ld + k0;
    v16bf f;
#pragma unroll
    for (int r = 0; r < 8; ++r) {
        const int kk = ((r & 4) ? 16 : 0) + khalf * 8 + (r & 3) * 2;
        unsigned int u = *(const unsigned int*)(p + kk);
        v2bf pr = __builtin_bit_cast(v2bf, u);
        f[2 * r]     = pr[0];
        f[2 * r + 1] = pr[1];
    }
    return f;
}

// ---------------------------------------------------------------------------
// 1) RMS norm over channel dim of x (B,C,T,H,W), write bf16 in (bt,c,h,w)
//    contiguous order == row-major A matrix (8192 x 384) for the QKV GEMM.
// ---------------------------------------------------------------------------
__global__ void rmsnorm_kernel(const float* __restrict__ x,
                               const float* __restrict__ gamma,
                               unsigned short* __restrict__ xn) {
    __shared__ float wsum[12];
    const int pix = blockIdx.x;              // bt*1024 + h*32 + w
    const int bt  = pix >> 10;
    const int hw  = pix & 1023;
    const int b   = bt >> 2;                 // T = 4
    const int t   = bt & 3;
    const int c   = threadIdx.x;             // 384 threads = 12 waves

    const size_t xi = (size_t)b * (CC * TT * HWSZ) + (size_t)c * (TT * HWSZ)
                    + (size_t)t * HWSZ + hw;
    const float v = x[xi];
    float ss = v * v;
#pragma unroll
    for (int off = 16; off > 0; off >>= 1) ss += __shfl_xor(ss, off, 32);
    const int lane = threadIdx.x & 31;
    const int wv   = threadIdx.x >> 5;
    if (lane == 0) wsum[wv] = ss;
    __syncthreads();
    float tot = 0.f;
#pragma unroll
    for (int i = 0; i < 12; ++i) tot += wsum[i];
    const float nrm   = sqrtf(tot);
    const float scale = SQRT_C / fmaxf(nrm, 1e-12f) * gamma[c];
    xn[(size_t)bt * (CC * HWSZ) + (size_t)c * HWSZ + hw] = f32_to_bf16_rne(v * scale);
}

// ---------------------------------------------------------------------------
// 2) f32 -> bf16 weight conversion
// ---------------------------------------------------------------------------
__global__ void cvt_bf16_kernel(const float* __restrict__ src,
                                unsigned short* __restrict__ dst, int n) {
    const int i = blockIdx.x * blockDim.x + threadIdx.x;
    if (i < n) dst[i] = f32_to_bf16_rne(src[i]);
}

// ---------------------------------------------------------------------------
// 3) QKV GEMM:  qkv[8192 x 1152] = A[8192 x 384] * W^T  (+bias, q-scale)
//    Wave tile 32M x 64N (8 wmma/K-step), block 8 waves (4M x 2N) = 128x128.
//    Register double buffer: next K-slab fragments in flight during WMMA.
// ---------------------------------------------------------------------------
__global__ void qkv_gemm_kernel(const unsigned short* __restrict__ A,
                                const unsigned short* __restrict__ Bw,
                                const float* __restrict__ bias,
                                float* __restrict__ out) {
    const int lane = threadIdx.x & 31;
    const int wv   = threadIdx.x >> 5;
    const int wm   = wv & 3;
    const int wn   = wv >> 2;
    const int m0   = blockIdx.y * 128 + wm * 32;
    const int n0   = blockIdx.x * 128 + wn * 64;

    v8f acc[2][4] = {};
    v16bf a0 = load_frag16x32(A, CC, m0,      0, lane);
    v16bf a1 = load_frag16x32(A, CC, m0 + 16, 0, lane);
    v16bf b[4];
#pragma unroll
    for (int nt = 0; nt < 4; ++nt) b[nt] = load_frag16x32(Bw, CC, n0 + nt * 16, 0, lane);

    for (int k0 = 0; k0 < CC; k0 += 32) {
        const int kn = (k0 + 32 < CC) ? k0 + 32 : 0;   // dummy wrap on last iter
        v16bf a0n = load_frag16x32(A, CC, m0,      kn, lane);
        v16bf a1n = load_frag16x32(A, CC, m0 + 16, kn, lane);
        v16bf bn[4];
#pragma unroll
        for (int nt = 0; nt < 4; ++nt) bn[nt] = load_frag16x32(Bw, CC, n0 + nt * 16, kn, lane);
#pragma unroll
        for (int nt = 0; nt < 4; ++nt) {
            acc[0][nt] = __builtin_amdgcn_wmma_f32_16x16x32_bf16(
                false, a0, false, b[nt], (short)0, acc[0][nt], false, false);
            acc[1][nt] = __builtin_amdgcn_wmma_f32_16x16x32_bf16(
                false, a1, false, b[nt], (short)0, acc[1][nt], false, false);
        }
        a0 = a0n; a1 = a1n;
#pragma unroll
        for (int nt = 0; nt < 4; ++nt) b[nt] = bn[nt];
    }

    const int mh = lane >> 4;
    const int nn = lane & 15;
#pragma unroll
    for (int mt = 0; mt < 2; ++mt) {
#pragma unroll
        for (int nt = 0; nt < 4; ++nt) {
            const int n = n0 + nt * 16 + nn;
            const float bv = bias[n];
#pragma unroll
            for (int r = 0; r < 8; ++r) {
                const int m = m0 + mt * 16 + r + 8 * mh;
                float v = acc[mt][nt][r] + bv;
                if (n < CC) v *= QSCALE;            // q * C^-0.5 folded in
                out[(size_t)m * N_QKV + n] = v;
            }
        }
    }
}

// ---------------------------------------------------------------------------
// 4) Neighborhood attention: one wave per query pixel, 49 keys.
//    q staged global->LDS with async b128 copies; float4 dot/AV loops;
//    softmax over 49 logits via shfl reductions.
// ---------------------------------------------------------------------------
__global__ void natten_kernel(const float* __restrict__ qkv,
                              const float* __restrict__ rpb,
                              unsigned short* __restrict__ aout) {
    __shared__ float4 q4[8][CC / 4];       // 96 float4 per wave
    __shared__ float  w_sh[8][52];
    const int lane = threadIdx.x & 31;
    const int wv   = threadIdx.x >> 5;
    const int pix  = blockIdx.x * 8 + wv;
    const int bt   = pix >> 10;
    const int hw   = pix & 1023;
    const int h    = hw >> 5;
    const int w    = hw & 31;

    // stage scaled q row (384 f32) into LDS via async copies (ASYNCcnt)
    const float4* qrow = (const float4*)(qkv + (size_t)pix * N_QKV);
#pragma unroll
    for (int r = 0; r < 3; ++r) {
        const int c4 = lane + r * 32;
        async_copy_b128(&q4[wv][c4], qrow + c4);
    }
    wait_asynccnt0();
    __syncthreads();

    const int sh = iclamp(h - KWIN / 2, 0, HH - KWIN);
    const int sw = iclamp(w - KWIN / 2, 0, WW - KWIN);

    float l0 = -3.0e38f, l1 = -3.0e38f;
#pragma unroll
    for (int rep = 0; rep < 2; ++rep) {
        const int j = lane + rep * 32;
        if (j < KWIN * KWIN) {
            const int ki = j / KWIN, kj = j % KWIN;
            const float4* k4 = (const float4*)(qkv +
                (size_t)(bt * HWSZ + (sh + ki) * WW + (sw + kj)) * N_QKV + CC);
            float dot = 0.f;
            for (int c = 0; c < CC / 4; ++c) {
                const float4 q = q4[wv][c];
                const float4 kk = k4[c];
                dot += q.x * kk.x + q.y * kk.y + q.z * kk.z + q.w * kk.w;
            }
            const int bh = ki - (h - sh) + (KWIN - 1);
            const int bw = kj - (w - sw) + (KWIN - 1);
            dot += rpb[bh * (2 * KWIN - 1) + bw];
            if (rep == 0) l0 = dot; else l1 = dot;
        }
    }
    // softmax over 49 logits spread across the wave
    float mx = fmaxf(l0, l1);
#pragma unroll
    for (int off = 16; off > 0; off >>= 1) mx = fmaxf(mx, __shfl_xor(mx, off, 32));
    const float e0 = __expf(l0 - mx);
    const float e1 = (lane < KWIN * KWIN - 32) ? __expf(l1 - mx) : 0.f;
    float s = e0 + e1;
#pragma unroll
    for (int off = 16; off > 0; off >>= 1) s += __shfl_xor(s, off, 32);
    const float inv = 1.0f / s;
    w_sh[wv][lane] = e0 * inv;
    if (lane < KWIN * KWIN - 32) w_sh[wv][lane + 32] = e1 * inv;
    __syncthreads();

    // attn @ V  (each lane owns 3 float4 channel groups = 12 channels)
#pragma unroll
    for (int r = 0; r < 3; ++r) {
        const int c4 = lane + r * 32;
        float ax = 0.f, ay = 0.f, az = 0.f, aw = 0.f;
        for (int j = 0; j < KWIN * KWIN; ++j) {
            const int ki = j / KWIN, kj = j % KWIN;
            const float4* v4 = (const float4*)(qkv +
                (size_t)(bt * HWSZ + (sh + ki) * WW + (sw + kj)) * N_QKV + 2 * CC);
            const float  wj = w_sh[wv][j];
            const float4 vv = v4[c4];
            ax += wj * vv.x; ay += wj * vv.y; az += wj * vv.z; aw += wj * vv.w;
        }
        const unsigned int lo = (unsigned int)f32_to_bf16_rne(ax)
                              | ((unsigned int)f32_to_bf16_rne(ay) << 16);
        const unsigned int hi = (unsigned int)f32_to_bf16_rne(az)
                              | ((unsigned int)f32_to_bf16_rne(aw) << 16);
        uint2 pk; pk.x = lo; pk.y = hi;
        ((uint2*)aout)[(size_t)pix * (CC / 4) + c4] = pk;
    }
}

// ---------------------------------------------------------------------------
// 5) Proj GEMM + bias + residual.  Row-major (m,n) flat index of the proj
//    output equals the (B,C,T,H,W) flat index of x/d_out (raw-view identity),
//    so the residual add folds directly into the epilogue.
// ---------------------------------------------------------------------------
__global__ void proj_gemm_kernel(const unsigned short* __restrict__ A,
                                 const unsigned short* __restrict__ Bw,
                                 const float* __restrict__ bias,
                                 const float* __restrict__ xres,
                                 float* __restrict__ out) {
    const int lane = threadIdx.x & 31;
    const int wv   = threadIdx.x >> 5;
    const int wm   = wv & 3;
    const int wn   = wv >> 2;
    const int m0   = blockIdx.y * 128 + wm * 32;
    const int n0   = blockIdx.x * 128 + wn * 64;

    v8f acc[2][4] = {};
    v16bf a0 = load_frag16x32(A, CC, m0,      0, lane);
    v16bf a1 = load_frag16x32(A, CC, m0 + 16, 0, lane);
    v16bf b[4];
#pragma unroll
    for (int nt = 0; nt < 4; ++nt) b[nt] = load_frag16x32(Bw, CC, n0 + nt * 16, 0, lane);

    for (int k0 = 0; k0 < CC; k0 += 32) {
        const int kn = (k0 + 32 < CC) ? k0 + 32 : 0;
        v16bf a0n = load_frag16x32(A, CC, m0,      kn, lane);
        v16bf a1n = load_frag16x32(A, CC, m0 + 16, kn, lane);
        v16bf bn[4];
#pragma unroll
        for (int nt = 0; nt < 4; ++nt) bn[nt] = load_frag16x32(Bw, CC, n0 + nt * 16, kn, lane);
#pragma unroll
        for (int nt = 0; nt < 4; ++nt) {
            acc[0][nt] = __builtin_amdgcn_wmma_f32_16x16x32_bf16(
                false, a0, false, b[nt], (short)0, acc[0][nt], false, false);
            acc[1][nt] = __builtin_amdgcn_wmma_f32_16x16x32_bf16(
                false, a1, false, b[nt], (short)0, acc[1][nt], false, false);
        }
        a0 = a0n; a1 = a1n;
#pragma unroll
        for (int nt = 0; nt < 4; ++nt) b[nt] = bn[nt];
    }

    const int mh = lane >> 4;
    const int nn = lane & 15;
#pragma unroll
    for (int mt = 0; mt < 2; ++mt) {
#pragma unroll
        for (int nt = 0; nt < 4; ++nt) {
            const int n = n0 + nt * 16 + nn;
            const float bv = bias[n];
#pragma unroll
            for (int r = 0; r < 8; ++r) {
                const int m = m0 + mt * 16 + r + 8 * mh;
                const size_t o = (size_t)m * CC + n;
                out[o] = acc[mt][nt][r] + bv + xres[o];
            }
        }
    }
}

// ---------------------------------------------------------------------------
// Launch
// ---------------------------------------------------------------------------
extern "C" void kernel_launch(void* const* d_in, const int* in_sizes, int n_in,
                              void* d_out, int out_size, void* d_ws, size_t ws_size,
                              hipStream_t stream) {
    const float* x      = (const float*)d_in[0];
    const float* gamma  = (const float*)d_in[1];
    const float* qkv_w  = (const float*)d_in[2];
    const float* qkv_b  = (const float*)d_in[3];
    const float* rpb    = (const float*)d_in[4];
    const float* proj_w = (const float*)d_in[5];
    const float* proj_b = (const float*)d_in[6];
    float* out = (float*)d_out;

    char* ws = (char*)d_ws;
    // workspace layout (bytes)
    unsigned short* xn       = (unsigned short*)(ws + 0);         // 8192*384*2  = 6,291,456
    unsigned short* qkvw_bf  = (unsigned short*)(ws + 6291456);   // 1152*384*2  =   884,736
    unsigned short* projw_bf = (unsigned short*)(ws + 7176192);   //  384*384*2  =   294,912
    float*          qkv      = (float*)(ws + 7471104);            // 8192*1152*4 = 37,748,736
    unsigned short* aout     = (unsigned short*)(ws + 45219840);  // 8192*384*2  = 6,291,456
    (void)ws_size; (void)in_sizes; (void)n_in; (void)out_size;

    rmsnorm_kernel<<<MROWS, CC, 0, stream>>>(x, gamma, xn);
    cvt_bf16_kernel<<<(N_QKV * CC + 255) / 256, 256, 0, stream>>>(qkv_w, qkvw_bf, N_QKV * CC);
    cvt_bf16_kernel<<<(CC * CC + 255) / 256, 256, 0, stream>>>(proj_w, projw_bf, CC * CC);
    qkv_gemm_kernel<<<dim3(N_QKV / 128, MROWS / 128), 256, 0, stream>>>(xn, qkvw_bf, qkv_b, qkv);
    natten_kernel<<<MROWS / 8, 256, 0, stream>>>(qkv, rpb, aout);
    proj_gemm_kernel<<<dim3(CC / 128, MROWS / 128), 256, 0, stream>>>(aout, projw_bf, proj_b, x, out);
}
